// MatchLSTMEncoder_34248069218413
// MI455X (gfx1250) — compile-verified, matrix operand order checked by reading, not run on hardware
//
#include <hip/hip_runtime.h>
#include <hip/hip_bf16.h>
#include <math.h>

// Problem dims (fixed by reference)
#define D      256
#define Qn     64
#define Pn     128
#define Bn     64
#define TWO_D  512
#define FOUR_D 1024
#define LDW_IH 33280   // (Q+1)*2D, leading dim of W_ih rows

typedef __attribute__((ext_vector_type(16))) __bf16 v16bf;
typedef __attribute__((ext_vector_type(8)))  float  v8f;
typedef __attribute__((ext_vector_type(4)))  int    i4v;

#define AS1 __attribute__((address_space(1)))
#define AS3 __attribute__((address_space(3)))

#if defined(__has_builtin)
#if __has_builtin(__builtin_amdgcn_global_load_async_to_lds_b128)
#define USE_ASYNC_LDS 1
#endif
#endif

__device__ __forceinline__ float sigmoidf_(float x) {
    return 1.0f / (1.0f + expf(-x));
}

// Stage one 16-byte chunk global -> LDS. Async (CDNA5 GLOBAL_LOAD_ASYNC_TO_LDS_B128,
// ASYNCcnt-tracked) when the builtin exists; plain LDS store fallback otherwise.
__device__ __forceinline__ void stage_b128(const float* g, float* l) {
#if USE_ASYNC_LDS
    __builtin_amdgcn_global_load_async_to_lds_b128(
        (AS1 i4v*)(g), (AS3 i4v*)(l), /*offset=*/0, /*cpol=*/0);
#else
    *(float4*)l = *(const float4*)g;
#endif
}

// Wait until at most `n` of this wave's async chunks are outstanding.
// Async loads complete in order, so waiting <=6 after issuing 6 chunks for the
// NEXT tile guarantees the PREVIOUS tile's 6 chunks have fully landed in LDS.
__device__ __forceinline__ void wait_async_le6() {
#if USE_ASYNC_LDS
    asm volatile("s_wait_asynccnt 0x6" ::: "memory");
#endif
}
__device__ __forceinline__ void wait_async_0() {
#if USE_ASYNC_LDS
    asm volatile("s_wait_asynccnt 0x0" ::: "memory");
#endif
}

// Build a bf16 16x32-layout fragment for one lane from an f32 LDS row.
// Documented 16-bit A/B layout: lanes 0-15 hold K = {0..7, 16..23},
// lanes 16-31 hold K = {8..15, 24..31}  (kh = 0 or 8).
// -> four aligned float4 LDS reads (ds_load_b128) + packed cvt to bf16.
__device__ __forceinline__ v16bf frag_from_lds(const float* rowp, int kh) {
    float4 a = *(const float4*)(rowp + kh);
    float4 b = *(const float4*)(rowp + kh + 4);
    float4 c = *(const float4*)(rowp + 16 + kh);
    float4 d = *(const float4*)(rowp + 20 + kh);
    v16bf v;
    v[0]  = (__bf16)a.x; v[1]  = (__bf16)a.y; v[2]  = (__bf16)a.z; v[3]  = (__bf16)a.w;
    v[4]  = (__bf16)b.x; v[5]  = (__bf16)b.y; v[6]  = (__bf16)b.z; v[7]  = (__bf16)b.w;
    v[8]  = (__bf16)c.x; v[9]  = (__bf16)c.y; v[10] = (__bf16)c.z; v[11] = (__bf16)c.w;
    v[12] = (__bf16)d.x; v[13] = (__bf16)d.y; v[14] = (__bf16)d.z; v[15] = (__bf16)d.w;
    return v;
}

// ---------------------------------------------------------------------------
// bf16 WMMA GEMM:  C[M,N] = A[M,K] * W[N,K]^T (+bias1+bias2)
// Block = 256 threads = 8 waves (2 waves in M x 4 in N), tile 64(M) x 128(N).
// Each wave: 2x2 grid of 16x16 accumulators -> 4 WMMAs per 32-wide K step.
// Double-buffered async global->LDS staging (6 b128 chunks per thread per
// tile) hides global latency behind WMMA + cvt.  LDS = 2 x 24 KB.
// Batching over blockIdx.z: A += z*aBatch, W col-offset = wOff0 + z*wOffStep,
// C += z*cBatch.  Requires M%64==0, N%128==0, K%32==0.
// ---------------------------------------------------------------------------
__global__ void __launch_bounds__(256)
k_gemm_bf16(const float* __restrict__ A, int lda, long aBatch,
            const float* __restrict__ W, int ldw, long wOff0, long wOffStep,
            const float* __restrict__ bias1, const float* __restrict__ bias2,
            float* __restrict__ C, int ldc, long cBatch, int K)
{
    __shared__ float lA[2][64 * 32];    // 2 x  8 KB
    __shared__ float lB[2][128 * 32];   // 2 x 16 KB

    const int tid   = threadIdx.x;
    const int wave  = tid >> 5;
    const int lane  = tid & 31;
    const int waveM = wave >> 2;     // 0..1
    const int waveN = wave & 3;      // 0..3
    const int mTile = blockIdx.x * 64;
    const int nTile = blockIdx.y * 128;
    const long z    = blockIdx.z;

    const float* Ab = A + z * aBatch + (long)mTile * lda;
    const float* Wp = W + wOff0 + z * wOffStep;
    float*       Cb = C + z * cBatch + (long)mTile * ldc + nTile;

    v8f acc[2][2] = {};

    const int rsub = lane & 15;        // row within a 16-wide subtile
    const int kh   = (lane >> 4) * 8;  // K-half select per layout

    // Issue one tile's staging: A 64x32 (2 chunks/thread) + B 128x32 (4/thread)
    auto stage_tile = [&](int buf, int k0) {
#pragma unroll
        for (int j = 0; j < 2; ++j) {
            int i = (tid + j * 256) * 4;
            int r = i >> 5, cc = i & 31;
            stage_b128(Ab + (long)r * lda + k0 + cc, &lA[buf][i]);
        }
#pragma unroll
        for (int j = 0; j < 4; ++j) {
            int i = (tid + j * 256) * 4;
            int r = i >> 5, cc = i & 31;
            stage_b128(Wp + (long)(nTile + r) * ldw + k0 + cc, &lB[buf][i]);
        }
    };

    const int ntiles = K >> 5;
    stage_tile(0, 0);                      // prologue

    for (int it = 0; it < ntiles; ++it) {
        const int cur = it & 1;
        if (it + 1 < ntiles) {
            stage_tile((it + 1) & 1, (it + 1) * 32);  // prefetch next tile
            wait_async_le6();                         // current tile landed
        } else {
            wait_async_0();
        }
        __syncthreads();   // all waves' writes for tile `it` visible

        v16bf af[2], bfr[2];
#pragma unroll
        for (int m = 0; m < 2; ++m)
            af[m] = frag_from_lds(&lA[cur][((waveM * 2 + m) * 16 + rsub) * 32], kh);
#pragma unroll
        for (int n = 0; n < 2; ++n)
            bfr[n] = frag_from_lds(&lB[cur][((waveN * 2 + n) * 16 + rsub) * 32], kh);

#pragma unroll
        for (int m = 0; m < 2; ++m)
#pragma unroll
            for (int n = 0; n < 2; ++n)
                acc[m][n] = __builtin_amdgcn_wmma_f32_16x16x32_bf16(
                    /*neg_a=*/false, af[m], /*neg_b=*/false, bfr[n],
                    /*c_mod=*/(short)0, acc[m][n],
                    /*reuse_a=*/false, /*reuse_b=*/false);

        __syncthreads();   // tile `it` fully consumed before its buffer reuse
    }

    // C layout: lanes 0-15 -> N=lane, M=r; lanes 16-31 -> N=lane-16, M=r+8
    const int col   = lane & 15;
    const int rbase = (lane >> 4) * 8;
#pragma unroll
    for (int m = 0; m < 2; ++m) {
        const int mrow = (waveM * 2 + m) * 16 + rbase;
#pragma unroll
        for (int n = 0; n < 2; ++n) {
            const int ncol = (waveN * 2 + n) * 16 + col;
            float bsum = 0.0f;
            if (bias1) bsum += bias1[nTile + ncol];
            if (bias2) bsum += bias2[nTile + ncol];
#pragma unroll
            for (int r = 0; r < 8; ++r)
                Cb[(long)(mrow + r) * ldc + ncol] = acc[m][n][r] + bsum;
        }
    }
}

// ---------------------------------------------------------------------------
// Per-step attention: s = s_a[t,b,:] + h@Wb^T ; logits[q] = alpha_w . tanh(ga+s)
// softmax over Q -> alpha[b,:].  One block per batch element, 256 threads.
// ---------------------------------------------------------------------------
__global__ void __launch_bounds__(256)
k_attn(int t, const float* __restrict__ sa, const float* __restrict__ ga,
       const float* __restrict__ Wb, const float* __restrict__ alpha_w,
       const float* __restrict__ alpha_b, const float* __restrict__ h,
       float* __restrict__ alpha)
{
    const int b   = blockIdx.x;
    const int tid = threadIdx.x;

    __shared__ float hb[D], ssh[D], aw[D], lg[Qn];
    __shared__ float red;

    hb[tid] = h[b * D + tid];
    aw[tid] = alpha_w[tid];
    __syncthreads();

    // s[tid] = s_a[t,b,tid] + sum_e h[b,e] * Wb[tid,e]
    float acc = sa[(((long)t * Bn) + b) * D + tid];
    const float* wrow = Wb + (long)tid * D;
#pragma unroll 4
    for (int e = 0; e < D; ++e) acc += hb[e] * wrow[e];
    ssh[tid] = acc;
    __syncthreads();

    // 8 waves x 8 q each; per q lane covers 8 d's, wave-reduce (wave32)
    const int wave = tid >> 5, lane = tid & 31;
    for (int qi = 0; qi < 8; ++qi) {
        int q = wave * 8 + qi;
        const float* gq = ga + ((long)q * Bn + b) * D;
        float p = 0.0f;
#pragma unroll
        for (int j = 0; j < 8; ++j) {
            int d = lane + j * 32;
            p += aw[d] * tanhf(gq[d] + ssh[d]);
        }
        p += __shfl_xor(p, 16, 32);
        p += __shfl_xor(p,  8, 32);
        p += __shfl_xor(p,  4, 32);
        p += __shfl_xor(p,  2, 32);
        p += __shfl_xor(p,  1, 32);
        if (lane == 0) lg[q] = p + alpha_b[0];
    }
    __syncthreads();

    if (tid == 0) {
        float mx = lg[0];
        for (int q = 1; q < Qn; ++q) mx = fmaxf(mx, lg[q]);
        float s = 0.0f;
        for (int q = 0; q < Qn; ++q) { float e = expf(lg[q] - mx); lg[q] = e; s += e; }
        red = s;
    }
    __syncthreads();
    if (tid < Qn) alpha[b * Qn + tid] = lg[tid] / red;
}

// ---------------------------------------------------------------------------
// Per-step LSTM: gates = pre_ih[t] + sum_q alpha*U + h@W_hh^T, then pointwise.
// One block per batch element, 256 threads (thread = hidden dim d).
// W_hh (1 MB) and U (16 MB) are L2-resident on MI455X (192 MB L2).
// ---------------------------------------------------------------------------
__global__ void __launch_bounds__(256)
k_step(int t, const float* __restrict__ preih, const float* __restrict__ U,
       const float* __restrict__ W_hh, const float* __restrict__ alpha,
       float* __restrict__ h, float* __restrict__ c, float* __restrict__ out)
{
    const int b = blockIdx.x;
    const int d = threadIdx.x;

    __shared__ float hb[D], al[Qn];
    hb[d] = h[b * D + d];
    if (d < Qn) al[d] = alpha[b * Qn + d];
    __syncthreads();

    float g4[4];
#pragma unroll
    for (int g = 0; g < 4; ++g) {
        int k = g * D + d;
        float acc = preih[(((long)t * Bn) + b) * FOUR_D + k];  // includes b_ih+b_hh
        const float* urow = U + (long)b * FOUR_D + k;          // U[q,b,k]
#pragma unroll 4
        for (int q = 0; q < Qn; ++q)
            acc += al[q] * urow[(long)q * Bn * FOUR_D];
        const float* wrow = W_hh + (long)k * D;
        if (g < 3)  // prefetch next gate's W_hh row (global_prefetch_b8)
            __builtin_prefetch(W_hh + (long)(k + D) * D, 0, 1);
#pragma unroll 4
        for (int e = 0; e < D; ++e) acc += hb[e] * wrow[e];
        g4[g] = acc;
    }

    // PyTorch gate order: i, f, g, o
    float i_g = sigmoidf_(g4[0]);
    float f_g = sigmoidf_(g4[1]);
    float gg  = tanhf(g4[2]);
    float o_g = sigmoidf_(g4[3]);
    float cn  = f_g * c[b * D + d] + i_g * gg;
    float hn  = o_g * tanhf(cn);
    c[b * D + d] = cn;
    h[b * D + d] = hn;
    out[(((long)t * Bn) + b) * D + d] = hn;
}

__global__ void k_init(const float* __restrict__ h0, const float* __restrict__ c0,
                       float* __restrict__ h, float* __restrict__ c)
{
    int i = blockIdx.x * blockDim.x + threadIdx.x;
    h[i] = h0[i];
    c[i] = c0[i];
}

// ---------------------------------------------------------------------------
extern "C" void kernel_launch(void* const* d_in, const int* in_sizes, int n_in,
                              void* d_out, int out_size, void* d_ws, size_t ws_size,
                              hipStream_t stream)
{
    (void)in_sizes; (void)n_in; (void)out_size; (void)ws_size;

    const float* H_p     = (const float*)d_in[0];
    const float* h_ri    = (const float*)d_in[1];
    const float* H_q     = (const float*)d_in[2];
    const float* hidden  = (const float*)d_in[3];
    const float* Wa      = (const float*)d_in[4];
    const float* ba      = (const float*)d_in[5];
    const float* Wb      = (const float*)d_in[6];
    const float* Wg      = (const float*)d_in[7];
    const float* alpha_w = (const float*)d_in[8];
    const float* alpha_b = (const float*)d_in[9];
    const float* W_ih    = (const float*)d_in[10];
    const float* W_hh    = (const float*)d_in[11];
    const float* b_ih    = (const float*)d_in[12];
    const float* b_hh    = (const float*)d_in[13];
    float* out = (float*)d_out;

    // Workspace layout (floats); total ~63 MB (fits MI455X 192 MB L2)
    float* ws    = (float*)d_ws;
    float* ga    = ws;                                  // Q*B*D
    float* sa    = ga    + (long)Qn * Bn * D;           // P*B*D
    float* preih = sa    + (long)Pn * Bn * D;           // P*B*4D
    float* U     = preih + (long)Pn * Bn * FOUR_D;      // Q*B*4D
    float* h     = U     + (long)Qn * Bn * FOUR_D;      // B*D
    float* c     = h     + Bn * D;                      // B*D
    float* alpha = c     + Bn * D;                      // B*Q

    // --- Hoisted WMMA GEMMs (time-independent) ---
    // ga = H_q @ Wg^T                         M=4096, N=256, K=512
    k_gemm_bf16<<<dim3((Qn * Bn) / 64, D / 128, 1), 256, 0, stream>>>(
        H_q, TWO_D, 0, Wg, TWO_D, 0, 0, nullptr, nullptr, ga, D, 0, TWO_D);
    // s_a = H_p @ Wa^T + ba                   M=8192, N=256, K=512
    k_gemm_bf16<<<dim3((Pn * Bn) / 64, D / 128, 1), 256, 0, stream>>>(
        H_p, TWO_D, 0, Wa, TWO_D, 0, 0, ba, nullptr, sa, D, 0, TWO_D);
    // pre_ih = H_p @ W_ih[:, :2D]^T + b_ih + b_hh   M=8192, N=1024, K=512
    k_gemm_bf16<<<dim3((Pn * Bn) / 64, FOUR_D / 128, 1), 256, 0, stream>>>(
        H_p, TWO_D, 0, W_ih, LDW_IH, 0, 0, b_ih, b_hh, preih, FOUR_D, 0, TWO_D);
    // U[q] = H_q[q] @ W_ih[:, (q+1)*2D:(q+2)*2D]^T  batched over q (z-dim)
    k_gemm_bf16<<<dim3(Bn / 64, FOUR_D / 128, Qn), 256, 0, stream>>>(
        H_q, TWO_D, (long)Bn * TWO_D, W_ih, LDW_IH, TWO_D, TWO_D,
        nullptr, nullptr, U, FOUR_D, (long)Bn * FOUR_D, TWO_D);

    // --- Init recurrent state ---
    k_init<<<Bn, D, 0, stream>>>(h_ri, hidden, h, c);

    // --- Sequential recurrence: 128 steps x (attention, lstm) ---
    for (int t = 0; t < Pn; ++t) {
        k_attn<<<Bn, 256, 0, stream>>>(t, sa, ga, Wb, alpha_w, alpha_b, h, alpha);
        k_step<<<Bn, 256, 0, stream>>>(t, preih, U, W_hh, alpha, h, c, out);
    }
}
